// DeepSeekNativeEPWrapper_41987600286210
// MI455X (gfx1250) — matrix operand order repaired and checked by reading, not compiled, not run on hardware
//
#include <hip/hip_runtime.h>
#include <cstddef>

// ---------------- types ----------------
typedef __attribute__((ext_vector_type(16))) __bf16 v16bf;
typedef __attribute__((ext_vector_type(8)))  float  v8f;
typedef __attribute__((ext_vector_type(4))) unsigned int v4u;
typedef __attribute__((ext_vector_type(4)))  float  v4f;
typedef __attribute__((ext_vector_type(2))) unsigned int v2u;

struct Q2 { v4u a, b; };

static __device__ __forceinline__ v16bf frag_cast(v4u a, v4u b) {
  Q2 q{a, b};
  return __builtin_bit_cast(v16bf, q);
}

static __device__ __forceinline__ unsigned short f2bf(float f) {
  unsigned u = __builtin_bit_cast(unsigned, f);
  unsigned r = u + 0x7FFFu + ((u >> 16) & 1u);   // round-to-nearest-even
  return (unsigned short)(r >> 16);
}

static __device__ __forceinline__ v8f zero8() {
  v8f z = {0.f, 0.f, 0.f, 0.f, 0.f, 0.f, 0.f, 0.f};
  return z;
}

// fast silu: g * sigmoid(g) with raw v_rcp_f32 instead of IEEE divide expansion
static __device__ __forceinline__ float fast_silu(float g) {
#if defined(__has_builtin) && __has_builtin(__builtin_amdgcn_rcpf)
  return g * __builtin_amdgcn_rcpf(1.f + __expf(-g));
#else
  return g / (1.f + __expf(-g));
#endif
}

// ---------------- problem constants ----------------
#define T_TOK 4096
#define DDIM  1024
#define NEXP  8
#define FDIM  512
#define FSDIM 1024
#define NCAT  5120   // 8*512 + 1024 concatenated hidden columns

#define BM 128
#define BN 128
#define KT 32
#define LDSK 40      // padded LDS row stride (bf16 units): conflict-free, 16B-aligned chunks

// ---------------- optional async global->LDS path (probe via __has_builtin) ----------------
#if defined(__has_builtin)
#if __has_builtin(__builtin_amdgcn_global_load_async_to_lds_b128) && \
    __has_builtin(__builtin_amdgcn_s_wait_asynccnt)
#define USE_ASYNC_LDS 1
#endif
#endif
#ifndef USE_ASYNC_LDS
#define USE_ASYNC_LDS 0
#endif

#if USE_ASYNC_LDS
// Builtin signature (from compiler diagnostic): params are pointers to
// 'int __attribute__((vector_size(16)))' in AS1 (global) / AS3 (LDS).
typedef int gvec4i __attribute__((vector_size(16)));
typedef __attribute__((address_space(1))) gvec4i* as1_v4i_p;
typedef __attribute__((address_space(3))) gvec4i* as3_v4i_p;
static __device__ __forceinline__ void async_cp16(const unsigned short* g, unsigned short* l) {
  __builtin_amdgcn_global_load_async_to_lds_b128((as1_v4i_p)g, (as3_v4i_p)l, 0, 0);
}
#endif

// ---------------- tile movement helpers ----------------
// Both A and B operands live in memory as [rows][K] with K contiguous
// (B is pre-transposed by the tcvt kernel), so one helper serves both.
static __device__ __forceinline__ void load_tile(const unsigned short* __restrict__ A,
                                                 int ld, int row0, int k0,
                                                 v4u& r0, v4u& r1, int tid) {
  int m = tid >> 1, h = (tid & 1) << 4;
  const v4u* s = (const v4u*)(A + (size_t)(row0 + m) * ld + k0 + h);
  r0 = s[0]; r1 = s[1];
}
static __device__ __forceinline__ void store_tile(unsigned short (*Ls)[LDSK],
                                                  v4u r0, v4u r1, int tid) {
  int m = tid >> 1, h = (tid & 1) << 4;
  v4u* d = (v4u*)&Ls[m][h];
  d[0] = r0; d[1] = r1;
}
#if USE_ASYNC_LDS
static __device__ __forceinline__ void async_tile(const unsigned short* __restrict__ A,
                                                  int ld, int row0, int k0,
                                                  unsigned short (*Ls)[LDSK], int tid) {
  int m = tid >> 1, h = (tid & 1) << 4;
  const unsigned short* g = A + (size_t)(row0 + m) * ld + k0 + h;
  async_cp16(g,     &Ls[m][h]);
  async_cp16(g + 8, &Ls[m][h + 8]);
}
#endif

// A fragment: 16x32 bf16, lanes 0-15 rows M0-15 K{0..7,16..23}; lanes 16-31 K{8..15,24..31}
static __device__ __forceinline__ v16bf load_fragA(const unsigned short (*As)[LDSK],
                                                   int mbase, int lane) {
  int m = mbase + (lane & 15);
  int c0 = (lane < 16) ? 0 : 8;
  v4u q0 = *(const v4u*)&As[m][c0];
  v4u q1 = *(const v4u*)&As[m][c0 + 16];
  return frag_cast(q0, q1);
}
// B fragment: 32x16 bf16, lane holds column n=lane%16, lanes 0-15 K{0..7,16..23}; lanes 16-31 K{8..15,24..31}
static __device__ __forceinline__ v16bf load_fragB(const unsigned short (*Bs)[LDSK],
                                                   int nbase, int lane) {
  int n = nbase + (lane & 15);
  int k0 = (lane < 16) ? 0 : 8;
  v4u q0 = *(const v4u*)&Bs[n][k0];
  v4u q1 = *(const v4u*)&Bs[n][k0 + 16];
  return frag_cast(q0, q1);
}

// ---------------- router: softmax + top-6 weights (others zeroed) ----------------
__global__ __launch_bounds__(256) void router_topk_kernel(const float* __restrict__ x,
                                                          const float* __restrict__ rw,
                                                          float* __restrict__ wt8) {
  int t = blockIdx.x * 256 + threadIdx.x;
  const float* xr = x + (size_t)t * DDIM;
  float acc[NEXP];
  #pragma unroll
  for (int e = 0; e < NEXP; ++e) acc[e] = 0.f;
  for (int d = 0; d < DDIM; ++d) {
    float xv = xr[d];
    const float* r = rw + d * NEXP;
    #pragma unroll
    for (int e = 0; e < NEXP; ++e) acc[e] += xv * r[e];
  }
  float mx = acc[0];
  #pragma unroll
  for (int e = 1; e < NEXP; ++e) mx = fmaxf(mx, acc[e]);
  float sum = 0.f;
  #pragma unroll
  for (int e = 0; e < NEXP; ++e) { acc[e] = __expf(acc[e] - mx); sum += acc[e]; }
  float inv = 1.f / sum;
  #pragma unroll
  for (int e = 0; e < NEXP; ++e) acc[e] *= inv;
  int i1 = -1, i2 = -1; float v1 = 1e30f, v2 = 1e30f;
  #pragma unroll
  for (int e = 0; e < NEXP; ++e) {
    float s = acc[e];
    if (s < v1) { v2 = v1; i2 = i1; v1 = s; i1 = e; }
    else if (s < v2) { v2 = s; i2 = e; }
  }
  float* o = wt8 + (size_t)t * NEXP;
  #pragma unroll
  for (int e = 0; e < NEXP; ++e) o[e] = (e == i1 || e == i2) ? 0.f : acc[e];
}

// ---------------- f32 -> bf16 conversion (4 elements / thread) ----------------
__global__ __launch_bounds__(256) void cvt_bf16_kernel(const float* __restrict__ src,
                                                       unsigned short* __restrict__ dst,
                                                       int n4) {
  int i = blockIdx.x * 256 + threadIdx.x;
  if (i < n4) {
    v4f f = ((const v4f*)src)[i];
    unsigned r0 = (unsigned)f2bf(f.x) | ((unsigned)f2bf(f.y) << 16);
    unsigned r1 = (unsigned)f2bf(f.z) | ((unsigned)f2bf(f.w) << 16);
    v2u o = {r0, r1};
    ((v2u*)dst)[i] = o;
  }
}

// ---------------- transpose + convert: dst[b][c][colOfs + ... ] = bf16(src[b][r][c]) ----
// src: [batch][R][C] f32 row-major.  dst rows indexed by c with leading stride ldD:
//   dst[b*R*C + c*ldD + colOfs + r] = bf16(src[b*R*C + r*C + c])
__global__ __launch_bounds__(256) void tcvt_kernel(const float* __restrict__ src,
                                                   unsigned short* __restrict__ dst,
                                                   int R, int C, int ldD, int colOfs) {
  __shared__ float tile[32][33];
  int b = blockIdx.z;
  const float* s = src + (size_t)b * R * C;
  unsigned short* d = dst + (size_t)b * R * C;   // batch>1 only used when ldD==R, colOfs==0
  int r0 = blockIdx.y * 32, c0 = blockIdx.x * 32;
  int tx = threadIdx.x & 31, ty = threadIdx.x >> 5;   // 32 x 8
  #pragma unroll
  for (int i = 0; i < 32; i += 8)
    tile[ty + i][tx] = s[(size_t)(r0 + ty + i) * C + c0 + tx];
  __syncthreads();
  #pragma unroll
  for (int i = 0; i < 32; i += 8)
    d[(size_t)(c0 + ty + i) * ldD + colOfs + r0 + tx] = f2bf(tile[tx][ty + i]);
}

// ---------------- fused gate+up GEMM: H = silu(X Wg) * (X Wu) * routeW ----------------
// X: [T,1024] bf16 row-major.  WgT/WuT: per-expert [N,1024] bf16 (pre-transposed, K contiguous).
// Output bf16 into hcat[T, 5120] at column (colbase + e*N).
__global__ __launch_bounds__(256) void gateup_kernel(const unsigned short* __restrict__ xb,
                                                     const unsigned short* __restrict__ wgT,
                                                     const unsigned short* __restrict__ wuT,
                                                     const float* __restrict__ wt8,
                                                     unsigned short* __restrict__ hcat,
                                                     int N, int colbase) {
  __shared__ __align__(16) unsigned short As[2][BM][LDSK];
  __shared__ __align__(16) unsigned short Bg[2][BN][LDSK];
  __shared__ __align__(16) unsigned short Bu[2][BN][LDSK];

  const int K = DDIM;
  int tid = threadIdx.x;
  int lane = tid & 31, wid = tid >> 5;
  int wm = (wid & 3) * 32, wn = (wid >> 2) * 64;
  int row0 = blockIdx.x * BM;
  int col0 = blockIdx.y * BN;
  int e = blockIdx.z;
  const unsigned short* Wg = wgT + (size_t)e * K * N;
  const unsigned short* Wu = wuT + (size_t)e * K * N;

  v8f accG[2][4], accU[2][4];
  #pragma unroll
  for (int i = 0; i < 2; ++i)
    #pragma unroll
    for (int j = 0; j < 4; ++j) { accG[i][j] = zero8(); accU[i][j] = zero8(); }

#if USE_ASYNC_LDS
  async_tile(xb, K, row0, 0, As[0], tid);
  async_tile(Wg, K, col0, 0, Bg[0], tid);
  async_tile(Wu, K, col0, 0, Bu[0], tid);
  __builtin_amdgcn_s_wait_asynccnt(0);
  __syncthreads();
#else
  v4u a0, a1, g0, g1, u0, u1;
  load_tile(xb, K, row0, 0, a0, a1, tid);
  load_tile(Wg, K, col0, 0, g0, g1, tid);
  load_tile(Wu, K, col0, 0, u0, u1, tid);
  store_tile(As[0], a0, a1, tid);
  store_tile(Bg[0], g0, g1, tid);
  store_tile(Bu[0], u0, u1, tid);
  __syncthreads();
#endif

  #pragma unroll 1
  for (int kt = 0; kt < K; kt += KT) {
    int cur = (kt / KT) & 1;
    int nxt = cur ^ 1;
    bool more = (kt + KT) < K;
#if USE_ASYNC_LDS
    if (more) {
      async_tile(xb, K, row0, kt + KT, As[nxt], tid);
      async_tile(Wg, K, col0, kt + KT, Bg[nxt], tid);
      async_tile(Wu, K, col0, kt + KT, Bu[nxt], tid);
    }
#else
    v4u a0, a1, g0, g1, u0, u1;
    if (more) {
      load_tile(xb, K, row0, kt + KT, a0, a1, tid);
      load_tile(Wg, K, col0, kt + KT, g0, g1, tid);
      load_tile(Wu, K, col0, kt + KT, u0, u1, tid);
    }
#endif
    v16bf af[2];
    #pragma unroll
    for (int mi = 0; mi < 2; ++mi) af[mi] = load_fragA(As[cur], wm + mi * 16, lane);
    #pragma unroll
    for (int ni = 0; ni < 4; ++ni) {
      v16bf bg = load_fragB(Bg[cur], wn + ni * 16, lane);
      v16bf bu = load_fragB(Bu[cur], wn + ni * 16, lane);
      #pragma unroll
      for (int mi = 0; mi < 2; ++mi) {
        accG[mi][ni] = __builtin_amdgcn_wmma_f32_16x16x32_bf16(
            false, af[mi], false, bg, (short)0, accG[mi][ni], false, false);
        accU[mi][ni] = __builtin_amdgcn_wmma_f32_16x16x32_bf16(
            false, af[mi], false, bu, (short)0, accU[mi][ni], false, false);
      }
    }
#if USE_ASYNC_LDS
    if (more) __builtin_amdgcn_s_wait_asynccnt(0);
    __syncthreads();
#else
    if (more) {
      store_tile(As[nxt], a0, a1, tid);
      store_tile(Bg[nxt], g0, g1, tid);
      store_tile(Bu[nxt], u0, u1, tid);
    }
    __syncthreads();
#endif
  }

  int colofs = colbase + e * N + col0;
  #pragma unroll
  for (int mi = 0; mi < 2; ++mi) {
    #pragma unroll
    for (int ni = 0; ni < 4; ++ni) {
      #pragma unroll
      for (int i = 0; i < 8; ++i) {
        int m = row0 + wm + mi * 16 + ((lane < 16) ? 0 : 8) + i;
        int n = colofs + wn + ni * 16 + (lane & 15);
        float g = accG[mi][ni][i], u = accU[mi][ni][i];
        float h = fast_silu(g) * u;
        float sc = wt8 ? wt8[(size_t)m * NEXP + e] : 1.f;
        hcat[(size_t)m * NCAT + n] = f2bf(h * sc);
      }
    }
  }
}

// ---------------- combined down-projection GEMM: out = Hcat @ WdcT^T (f32 out) -------
// hcat: [T,5120] bf16.  wdcT: [1024][5120] bf16 (pre-transposed, K contiguous).
__global__ __launch_bounds__(256) void down_kernel(const unsigned short* __restrict__ hcat,
                                                   const unsigned short* __restrict__ wdcT,
                                                   float* __restrict__ out) {
  __shared__ __align__(16) unsigned short As[2][BM][LDSK];
  __shared__ __align__(16) unsigned short Bs[2][BN][LDSK];

  const int K = NCAT;
  int tid = threadIdx.x;
  int lane = tid & 31, wid = tid >> 5;
  int wm = (wid & 3) * 32, wn = (wid >> 2) * 64;
  int row0 = blockIdx.x * BM;
  int col0 = blockIdx.y * BN;

  v8f acc[2][4];
  #pragma unroll
  for (int i = 0; i < 2; ++i)
    #pragma unroll
    for (int j = 0; j < 4; ++j) acc[i][j] = zero8();

#if USE_ASYNC_LDS
  async_tile(hcat, K, row0, 0, As[0], tid);
  async_tile(wdcT, K, col0, 0, Bs[0], tid);
  __builtin_amdgcn_s_wait_asynccnt(0);
  __syncthreads();
#else
  v4u a0, a1, b0, b1;
  load_tile(hcat, K, row0, 0, a0, a1, tid);
  load_tile(wdcT, K, col0, 0, b0, b1, tid);
  store_tile(As[0], a0, a1, tid);
  store_tile(Bs[0], b0, b1, tid);
  __syncthreads();
#endif

  #pragma unroll 1
  for (int kt = 0; kt < K; kt += KT) {
    int cur = (kt / KT) & 1;
    int nxt = cur ^ 1;
    bool more = (kt + KT) < K;
#if USE_ASYNC_LDS
    if (more) {
      async_tile(hcat, K, row0, kt + KT, As[nxt], tid);
      async_tile(wdcT, K, col0, kt + KT, Bs[nxt], tid);
    }
#else
    v4u a0, a1, b0, b1;
    if (more) {
      load_tile(hcat, K, row0, kt + KT, a0, a1, tid);
      load_tile(wdcT, K, col0, kt + KT, b0, b1, tid);
    }
#endif
    v16bf af[2];
    #pragma unroll
    for (int mi = 0; mi < 2; ++mi) af[mi] = load_fragA(As[cur], wm + mi * 16, lane);
    #pragma unroll
    for (int ni = 0; ni < 4; ++ni) {
      v16bf bf = load_fragB(Bs[cur], wn + ni * 16, lane);
      #pragma unroll
      for (int mi = 0; mi < 2; ++mi) {
        acc[mi][ni] = __builtin_amdgcn_wmma_f32_16x16x32_bf16(
            false, af[mi], false, bf, (short)0, acc[mi][ni], false, false);
      }
    }
#if USE_ASYNC_LDS
    if (more) __builtin_amdgcn_s_wait_asynccnt(0);
    __syncthreads();
#else
    if (more) {
      store_tile(As[nxt], a0, a1, tid);
      store_tile(Bs[nxt], b0, b1, tid);
    }
    __syncthreads();
#endif
  }

  #pragma unroll
  for (int mi = 0; mi < 2; ++mi) {
    #pragma unroll
    for (int ni = 0; ni < 4; ++ni) {
      #pragma unroll
      for (int i = 0; i < 8; ++i) {
        int m = row0 + wm + mi * 16 + ((lane < 16) ? 0 : 8) + i;
        int n = col0 + wn + ni * 16 + (lane & 15);
        out[(size_t)m * DDIM + n] = acc[mi][ni][i];
      }
    }
  }
}

// ---------------- host launch ----------------
extern "C" void kernel_launch(void* const* d_in, const int* in_sizes, int n_in,
                              void* d_out, int out_size, void* d_ws, size_t ws_size,
                              hipStream_t stream) {
  (void)in_sizes; (void)n_in; (void)out_size; (void)ws_size;
  const float* x        = (const float*)d_in[0];  // [2,2048,1024]
  const float* router_w = (const float*)d_in[1];  // [1024,8]
  const float* w_gate   = (const float*)d_in[2];  // [8,1024,512]
  const float* w_up     = (const float*)d_in[3];  // [8,1024,512]
  const float* w_down   = (const float*)d_in[4];  // [8,512,1024]
  const float* ws_gate  = (const float*)d_in[5];  // [1024,1024]
  const float* ws_up    = (const float*)d_in[6];  // [1024,1024]
  const float* ws_down  = (const float*)d_in[7];  // [1024,1024]
  float* out = (float*)d_out;

  char* p = (char*)d_ws;
  float* wt8 = (float*)p;            p += (size_t)T_TOK * NEXP * 4;
  unsigned short* xb   = (unsigned short*)p; p += (size_t)T_TOK * DDIM * 2;
  unsigned short* wgT  = (unsigned short*)p; p += (size_t)NEXP * DDIM * FDIM * 2;  // [E][F][D]
  unsigned short* wuT  = (unsigned short*)p; p += (size_t)NEXP * DDIM * FDIM * 2;  // [E][F][D]
  unsigned short* wsgT = (unsigned short*)p; p += (size_t)DDIM * FSDIM * 2;        // [FS][D]
  unsigned short* wsuT = (unsigned short*)p; p += (size_t)DDIM * FSDIM * 2;        // [FS][D]
  unsigned short* wdcT = (unsigned short*)p; p += (size_t)NCAT * DDIM * 2;         // [D][5120]
  unsigned short* hcat = (unsigned short*)p; p += (size_t)T_TOK * NCAT * 2;        // [T][5120]

  router_topk_kernel<<<T_TOK / 256, 256, 0, stream>>>(x, router_w, wt8);

  {
    int n4 = (int)((size_t)T_TOK * DDIM / 4);
    cvt_bf16_kernel<<<(n4 + 255) / 256, 256, 0, stream>>>(x, xb, n4);
  }
  // transpose-converts: dst[c*ldD + colOfs + r] = bf16(src[r*C + c]), batched over z
  tcvt_kernel<<<dim3(FDIM / 32, DDIM / 32, NEXP), 256, 0, stream>>>(w_gate, wgT, DDIM, FDIM, DDIM, 0);
  tcvt_kernel<<<dim3(FDIM / 32, DDIM / 32, NEXP), 256, 0, stream>>>(w_up,   wuT, DDIM, FDIM, DDIM, 0);
  tcvt_kernel<<<dim3(FSDIM / 32, DDIM / 32, 1),   256, 0, stream>>>(ws_gate, wsgT, DDIM, FSDIM, DDIM, 0);
  tcvt_kernel<<<dim3(FSDIM / 32, DDIM / 32, 1),   256, 0, stream>>>(ws_up,   wsuT, DDIM, FSDIM, DDIM, 0);
  // w_down flat [4096][1024] -> wdcT[:, 0:4096]; ws_down [1024][1024] -> wdcT[:, 4096:5120]
  tcvt_kernel<<<dim3(DDIM / 32, (NEXP * FDIM) / 32, 1), 256, 0, stream>>>(w_down, wdcT, NEXP * FDIM, DDIM, NCAT, 0);
  tcvt_kernel<<<dim3(DDIM / 32, FSDIM / 32, 1),         256, 0, stream>>>(ws_down, wdcT, FSDIM, DDIM, NCAT, NEXP * FDIM);

  // experts: 8 x (4096 x 512 x 1024), routing-weight-scaled into hcat cols [0,4096)
  dim3 ge(T_TOK / BM, FDIM / BN, NEXP);
  gateup_kernel<<<ge, 256, 0, stream>>>(xb, wgT, wuT, wt8, hcat, FDIM, 0);
  // shared expert: 4096 x 1024 x 1024, scale 1, into hcat cols [4096,5120)
  dim3 gs(T_TOK / BM, FSDIM / BN, 1);
  gateup_kernel<<<gs, 256, 0, stream>>>(xb, wsgT, wsuT, nullptr, hcat, FSDIM, NEXP * FDIM);
  // combined down projection: 4096 x 1024 x 5120 -> f32 output
  dim3 gd(T_TOK / BM, DDIM / BN, 1);
  down_kernel<<<gd, 256, 0, stream>>>(hcat, wdcT, out);
}
